// TripletMPNN_64055142253064
// MI455X (gfx1250) — matrix operand order repaired and checked by reading, not compiled, not run on hardware
//
#include <hip/hip_runtime.h>
#include <stddef.h>

// ---------------------------------------------------------------------------
// TripletMPNN for MI455X (gfx1250), wave32 + WMMA bf16 16x16x32, f32 accum.
// B=4, N=128, IN=OUT=EDIM=128, TF=8, E=8192.
// Round 2: vectorized LDS access — A/B fragments via ds_load_b128 (v8bf pairs),
// B pre-swizzled to fragment order in LDS, float4 epilogues, prefetch hints.
// ---------------------------------------------------------------------------

typedef __bf16 bf16_t;
typedef __attribute__((ext_vector_type(16))) __bf16 v16bf;
typedef __attribute__((ext_vector_type(8)))  __bf16 v8bf;
typedef __attribute__((ext_vector_type(4)))  __bf16 v4bf;
typedef __attribute__((ext_vector_type(8)))  float  v8f;

#define NEGINF (-3.0e38f)
#define NEGBIG (-1.0e9f)

// ---------------- workspace layout (float element offsets) -----------------
static constexpr size_t OFF_E      = 0;                       // 4*128*128*128
static constexpr size_t OFF_ADJ    = OFF_E      + 8388608;    // 4*128*128
static constexpr size_t OFF_MSG1   = OFF_ADJ    + 65536;      // 512*128
static constexpr size_t OFF_MSG2   = OFF_MSG1   + 65536;
static constexpr size_t OFF_MSGG   = OFF_MSG2   + 65536;      // 4*128
static constexpr size_t OFF_T1     = OFF_MSGG   + 512;        // 512*8
static constexpr size_t OFF_T2     = OFF_T1     + 4096;
static constexpr size_t OFF_T3     = OFF_T2     + 4096;
static constexpr size_t OFF_TG     = OFF_T3     + 4096;       // 4*8
static constexpr size_t OFF_TE1T   = OFF_TG     + 32;         // [b][j][i][8]
static constexpr size_t OFF_TE2T   = OFF_TE1T   + 524288;     // [b][k][i][8]
static constexpr size_t OFF_TRIRED = OFF_TE2T   + 524288;     // [b][j][k][8]
static constexpr size_t OFF_MSGF   = OFF_TRIRED + 524288;     // [b][i][j][128]
static constexpr size_t OFF_MSGMAX = OFF_MSGF   + 8388608;    // [b][j][128]
// total ~18.63M floats (~74.5 MB) — fits in the 192MB L2 of MI455X.

// ---------------- WMMA fragment helpers (ISA 7.12.2 layouts) ----------------
// A 16x32 bf16: lane<16 holds M=lane,   elems 0..7 = K k0..k0+7,   8..15 = K k0+16..+23
//               lane>=16 holds M=lane-16, elems 0..7 = K k0+8..+15, 8..15 = K k0+24..+31
// Rows are 16B aligned (lda=136 -> 272B) so each half is one ds_load_b128.
__device__ inline v16bf frag_a(const bf16_t* A, int lda, int m0, int k0, int lane) {
  const bf16_t* row = A + (m0 + (lane & 15)) * lda + k0 + ((lane >> 4) << 3);
  v8bf lo = *(const v8bf*)(row);
  v8bf hi = *(const v8bf*)(row + 16);
  return __builtin_shufflevector(lo, hi, 0, 1, 2, 3, 4, 5, 6, 7,
                                 8, 9, 10, 11, 12, 13, 14, 15);
}
// B pre-swizzled in LDS: [tile = kt*NT+nt][lane][16 bf16] contiguous per lane.
__device__ inline v16bf frag_b_sw(const bf16_t* Bfr, int tile, int lane) {
  const bf16_t* p = Bfr + (((tile << 5) + lane) << 4);
  v8bf lo = *(const v8bf*)(p);
  v8bf hi = *(const v8bf*)(p + 8);
  return __builtin_shufflevector(lo, hi, 0, 1, 2, 3, 4, 5, 6, 7,
                                 8, 9, 10, 11, 12, 13, 14, 15);
}
// C/D 16x16 f32: VGPR r -> M = r + 8*(lane>=16), N = lane&15.

// Stage W[128 x Nout] (global, f32) into LDS in fragment order.
// Slot s = (kt*NT + nt)*32 + lane; each slot = 16 contiguous bf16 (2 b128 stores).
template <int NT>
__device__ inline void stage_w_frag(const float* __restrict__ W, bf16_t* Bfr,
                                    int Nout, int tid) {
  const int nslots = 4 * NT * 32;
  for (int s = tid; s < nslots; s += 256) {
    const int lane = s & 31;
    const int nt = (s >> 5) % NT;
    const int kt = s / (NT * 32);
    const int k0 = (kt << 5) + ((lane >> 4) << 3);
    const int n = (nt << 4) + (lane & 15);
    v8bf lo = {}, hi = {};
    if (n < Nout) {
#pragma unroll
      for (int e = 0; e < 8; ++e) lo[e] = (bf16_t)W[(k0 + e) * Nout + n];
#pragma unroll
      for (int e = 0; e < 8; ++e) hi[e] = (bf16_t)W[(k0 + 16 + e) * Nout + n];
    }
    v8bf* dst = (v8bf*)(Bfr + (s << 4));
    dst[0] = lo;
    dst[1] = hi;
  }
}

// 128x128 slab GEMM: 8 waves, wave w owns rows [16w,16w+16), 8 col tiles, K=128.
template <bool ZERO>
__device__ inline void gemm_slab(const bf16_t (*Abf)[136], const bf16_t* Bfr,
                                 v8f* acc, int tid) {
  const int wave = tid >> 5, lane = tid & 31;
  const int m0 = wave << 4;
  if (ZERO) {
    v8f z = {};
#pragma unroll
    for (int nt = 0; nt < 8; ++nt) acc[nt] = z;
  }
#pragma unroll
  for (int kt = 0; kt < 4; ++kt) {
    v16bf a = frag_a(&Abf[0][0], 136, m0, kt * 32, lane);
#pragma unroll
    for (int nt = 0; nt < 8; ++nt) {
      v16bf b = frag_b_sw(Bfr, kt * 8 + nt, lane);
      acc[nt] = __builtin_amdgcn_wmma_f32_16x16x32_bf16(
          false, a, false, b, (short)0, acc[nt], false, false);
    }
  }
}

__device__ inline void store_tiles(float (*Cst)[132], const v8f* acc, int tid) {
  const int wave = tid >> 5, lane = tid & 31;
  const int n_ = lane & 15, mb = (wave << 4) + ((lane >> 4) << 3);
#pragma unroll
  for (int nt = 0; nt < 8; ++nt) {
    const int n = (nt << 4) + n_;
#pragma unroll
    for (int r = 0; r < 8; ++r) Cst[mb + r][n] = acc[nt][r];
  }
}

// LayerNorm(128 feats)+ReLU from Cst -> Abf(bf16). 2 threads/row, float4 reads,
// v4bf (b64) stores.
__device__ inline void ln_relu_to_abf(float (*Cst)[132], bf16_t (*Abf)[136],
                                      const float* __restrict__ sc,
                                      const float* __restrict__ of,
                                      float* redA, float* redB, int tid) {
  const int row = tid >> 1, fb = (tid & 1) << 6;
  float s = 0.f, ss = 0.f;
#pragma unroll 4
  for (int f = 0; f < 64; f += 4) {
    const float4 x4 = *(const float4*)(&Cst[row][fb + f]);
    s += x4.x + x4.y + x4.z + x4.w;
    ss += x4.x * x4.x + x4.y * x4.y + x4.z * x4.z + x4.w * x4.w;
  }
  redA[tid] = s;
  redB[tid] = ss;
  __syncthreads();
  const float st = redA[tid ^ 1] + s, sst = redB[tid ^ 1] + ss;
  const float mean = st * 0.0078125f;
  const float rstd = rsqrtf(sst * 0.0078125f - mean * mean + 1e-5f);
#pragma unroll 4
  for (int f = 0; f < 64; f += 4) {
    const float4 x4 = *(const float4*)(&Cst[row][fb + f]);
    const float4 s4 = *(const float4*)(sc + fb + f);
    const float4 o4 = *(const float4*)(of + fb + f);
    v4bf h;
    h[0] = (bf16_t)fmaxf(s4.x * rstd * (x4.x - mean) + o4.x, 0.f);
    h[1] = (bf16_t)fmaxf(s4.y * rstd * (x4.y - mean) + o4.y, 0.f);
    h[2] = (bf16_t)fmaxf(s4.z * rstd * (x4.z - mean) + o4.z, 0.f);
    h[3] = (bf16_t)fmaxf(s4.w * rstd * (x4.w - mean) + o4.w, 0.f);
    *(v4bf*)(&Abf[row][fb + f]) = h;
  }
  __syncthreads();
}

// stage a 128x128 f32 matrix (global) into Abf as bf16, b64 LDS stores
__device__ inline void stage_a128(const float* __restrict__ A, bf16_t (*Abf)[136],
                                  int tid) {
  for (int q = tid; q < 4096; q += 256) {
    const int row = q >> 5, c = (q & 31) << 2;
    const float4 v = *(const float4*)(A + ((size_t)row << 7) + c);
    v4bf h;
    h[0] = (bf16_t)v.x; h[1] = (bf16_t)v.y; h[2] = (bf16_t)v.z; h[3] = (bf16_t)v.w;
    *(v4bf*)(&Abf[row][c]) = h;
  }
}

// ------------------------------ kernels -------------------------------------

__global__ void zero_kernel(float4* __restrict__ p, size_t n4) {
  size_t idx = (size_t)blockIdx.x * blockDim.x + threadIdx.x;
  const size_t stride = (size_t)gridDim.x * blockDim.x;
  const float4 z = make_float4(0.f, 0.f, 0.f, 0.f);
  for (; idx < n4; idx += stride) p[idx] = z;
}

// one thread per (edge, feature): scatter-add edge_attr into dense e, mark adj
__global__ void scatter_kernel(const float* __restrict__ edge_attr,
                               const int* __restrict__ EI,
                               float* __restrict__ E, float* __restrict__ ADJ) {
  const int t = blockIdx.x * 256 + threadIdx.x;  // < 8192*128
  const int e = t >> 7, f = t & 127;
  const int s0 = EI[e], s1 = EI[8192 + e];
  const int b = s0 >> 7, src = s0 & 127, dst = s1 & 127;
  const size_t base = (((size_t)(b * 128 + src) * 128) + dst) << 7;
  atomicAdd(&E[base + f], edge_attr[((size_t)e << 7) + f]);
  if (f == 0) ADJ[((size_t)(b * 128 + src) << 7) + dst] = 1.0f;
}

// generic C[M x Nout] = A[M x 128] @ W[128 x Nout], NT = Nout tiles of 16 (8 or 1)
template <int NT>
__global__ __launch_bounds__(256) void gemm_k128_kernel(
    const float* __restrict__ A, const float* __restrict__ W,
    float* __restrict__ C, int M, int Nout) {
  __shared__ alignas(16) bf16_t Abf[128][136];
  __shared__ alignas(16) bf16_t Bfr[4 * NT * 32 * 16];
  const int tid = threadIdx.x;
  const int r0 = blockIdx.x * 128;
  for (int q = tid; q < 4096; q += 256) {
    const int row = q >> 5, c = (q & 31) << 2;
    const int gr = r0 + row;
    float4 v = make_float4(0.f, 0.f, 0.f, 0.f);
    if (gr < M) v = *(const float4*)(A + ((size_t)gr << 7) + c);
    v4bf h;
    h[0] = (bf16_t)v.x; h[1] = (bf16_t)v.y; h[2] = (bf16_t)v.z; h[3] = (bf16_t)v.w;
    *(v4bf*)(&Abf[row][c]) = h;
  }
  stage_w_frag<NT>(W, Bfr, Nout, tid);
  __syncthreads();

  const int wave = tid >> 5, lane = tid & 31;
  const int m0 = wave << 4;
  v8f acc[NT];
  {
    v8f z = {};
#pragma unroll
    for (int nt = 0; nt < NT; ++nt) acc[nt] = z;
  }
#pragma unroll
  for (int kt = 0; kt < 4; ++kt) {
    v16bf a = frag_a(&Abf[0][0], 136, m0, kt * 32, lane);
#pragma unroll
    for (int nt = 0; nt < NT; ++nt) {
      v16bf b = frag_b_sw(Bfr, kt * NT + nt, lane);
      acc[nt] = __builtin_amdgcn_wmma_f32_16x16x32_bf16(
          false, a, false, b, (short)0, acc[nt], false, false);
    }
  }
  const int n_ = lane & 15, mb = m0 + ((lane >> 4) << 3);
#pragma unroll
  for (int nt = 0; nt < NT; ++nt) {
    const int n = nt * 16 + n_;
    if (n < Nout) {
#pragma unroll
      for (int r = 0; r < 8; ++r) {
        const int gr = r0 + mb + r;
        if (gr < M) C[(size_t)gr * Nout + n] = acc[nt][r];
      }
    }
  }
}

// Fused MPNN slab: block = (b,i). Stages e[b,i] once in LDS; computes te1/te2
// projections, then the 3-GEMM LN/ReLU chain entirely on-chip; emits masked msgs.
__global__ __launch_bounds__(256) void mpnn_slab_kernel(
    const float* __restrict__ E, const float* __restrict__ ADJ,
    const float* __restrict__ MSG1, const float* __restrict__ MSG2,
    const float* __restrict__ MSGG,
    const float* __restrict__ W_me, const float* __restrict__ W_mlp1,
    const float* __restrict__ W_mlp2,
    const float* __restrict__ ln1_s, const float* __restrict__ ln1_o,
    const float* __restrict__ ln2_s, const float* __restrict__ ln2_o,
    const float* __restrict__ W_te1, const float* __restrict__ W_te2,
    float* __restrict__ TE1T, float* __restrict__ TE2T,
    float* __restrict__ MSGF) {
  __shared__ alignas(16) bf16_t Abf[128][136];   // 34 KB
  __shared__ alignas(16) bf16_t Bfr[16384];      // 32 KB (4x8 tiles x 32 x 16)
  __shared__ alignas(16) float  Cst[128][132];   // 66 KB
  __shared__ float adjS[128], m2S[128], mgS[128];
  __shared__ float redA[256], redB[256];

  const int tid = threadIdx.x;
  const int bi = blockIdx.x;  // b*128 + i
  const int b = bi >> 7, i = bi & 127;

  // prefetch downstream weights into cache while stage 1 runs (global_prefetch_b8)
  __builtin_prefetch(W_mlp1 + ((tid & 63) << 8), 0, 1);
  __builtin_prefetch(W_mlp2 + ((tid & 63) << 8), 0, 1);

  stage_a128(E + ((size_t)bi << 14), Abf, tid);
  stage_w_frag<8>(W_me, Bfr, 128, tid);
  if (tid < 128) {
    adjS[tid] = ADJ[((size_t)bi << 7) + tid];
    m2S[tid] = MSG2[((size_t)bi << 7) + tid];
    mgS[tid] = MSGG[(b << 7) + tid];
  }
  __syncthreads();

  // te1[b,i,j,:], te2[b,i,j,:] = e[b,i,j,:] @ W_te{1,2}; store transposed [b][j][i][8]
  {
    const int j = tid >> 1, fb = (tid & 1) << 2;
    float a1[4] = {0.f, 0.f, 0.f, 0.f}, a2[4] = {0.f, 0.f, 0.f, 0.f};
    for (int c = 0; c < 128; c += 8) {
      const v8bf av8 = *(const v8bf*)(&Abf[j][c]);
#pragma unroll
      for (int u = 0; u < 8; ++u) {
        const float av = (float)av8[u];
#pragma unroll
        for (int q = 0; q < 4; ++q) {
          a1[q] += av * W_te1[((c + u) << 3) + fb + q];
          a2[q] += av * W_te2[((c + u) << 3) + fb + q];
        }
      }
    }
    const size_t o = (((size_t)(b * 128 + j) << 7) + i) * 8 + fb;
    *(float4*)(TE1T + o) = make_float4(a1[0], a1[1], a1[2], a1[3]);
    *(float4*)(TE2T + o) = make_float4(a2[0], a2[1], a2[2], a2[3]);
  }

  v8f acc[8];
  // stage 1: msg_e = e @ W_me; then masked add of msg1[j] + msg2[i] + msg_g
  gemm_slab<true>(Abf, Bfr, acc, tid);
  store_tiles(Cst, acc, tid);
  __syncthreads();
  for (int q = tid; q < 4096; q += 256) {
    const int m = q >> 5, c = (q & 31) << 2;
    float4 v = *(const float4*)(&Cst[m][c]);
    const float am = adjS[m];
    const float4 g = *(const float4*)(MSG1 + ((size_t)(b * 128 + m) << 7) + c);
    v.x = v.x * am + g.x + m2S[c] + mgS[c];
    v.y = v.y * am + g.y + m2S[c + 1] + mgS[c + 1];
    v.z = v.z * am + g.z + m2S[c + 2] + mgS[c + 2];
    v.w = v.w * am + g.w + m2S[c + 3] + mgS[c + 3];
    *(float4*)(&Cst[m][c]) = v;
  }
  __syncthreads();

  // stage 2: LN -> ReLU -> @W_mlp1
  ln_relu_to_abf(Cst, Abf, ln1_s, ln1_o, redA, redB, tid);
  stage_w_frag<8>(W_mlp1, Bfr, 128, tid);
  __syncthreads();
  gemm_slab<true>(Abf, Bfr, acc, tid);
  store_tiles(Cst, acc, tid);
  __syncthreads();

  // stage 3: LN -> ReLU -> @W_mlp2 -> masked coalesced write for max-reduction
  ln_relu_to_abf(Cst, Abf, ln2_s, ln2_o, redA, redB, tid);
  stage_w_frag<8>(W_mlp2, Bfr, 128, tid);
  __syncthreads();
  gemm_slab<true>(Abf, Bfr, acc, tid);
  store_tiles(Cst, acc, tid);
  __syncthreads();
  {
    float* outp = MSGF + ((size_t)bi << 14);
    const float4 neg = make_float4(NEGBIG, NEGBIG, NEGBIG, NEGBIG);
    for (int q = tid; q < 4096; q += 256) {
      const int m = q >> 5, c = (q & 31) << 2;
      float4 v = *(const float4*)(&Cst[m][c]);
      if (adjS[m] == 0.f) v = neg;
      *(float4*)(outp + (m << 7) + c) = v;
    }
  }
}

// max over i of masked msgs: block = (b,j)
__global__ __launch_bounds__(256) void maxi_kernel(const float* __restrict__ MSGF,
                                                   float* __restrict__ MM) {
  __shared__ float red[128];
  const int tid = threadIdx.x;
  const int bj = blockIdx.x;
  const int b = bj >> 7, j = bj & 127;
  const int f = tid & 127, ih = tid >> 7;
  float acc = NEGINF;
  for (int i = ih; i < 128; i += 2)
    acc = fmaxf(acc, MSGF[(((size_t)(b * 128 + i) << 7) + j) * 128 + f]);
  if (ih == 1) red[f] = acc;
  __syncthreads();
  if (ih == 0) MM[((size_t)bj << 7) + f] = fmaxf(acc, red[f]);
}

// tri_red[b,j,k,f] = max_i (t1[b,i,f] + te1[b,i,j,f] + te2[b,i,k,f]); block=(b,j)
__global__ __launch_bounds__(256) void trired_kernel(
    const float* __restrict__ T1, const float* __restrict__ TE1T,
    const float* __restrict__ TE2T, float* __restrict__ TRIRED) {
  __shared__ alignas(16) float s1[128][8];
  const int tid = threadIdx.x;
  const int bj = blockIdx.x;
  const int b = bj >> 7;
  for (int q = tid; q < 1024; q += 256) {
    const int i = q >> 3, f = q & 7;
    s1[i][f] = T1[((size_t)(b * 128 + i) << 3) + f] +
               TE1T[(((size_t)bj << 7) + i) * 8 + f];
  }
  __syncthreads();
  const int wave = tid >> 5, lane = tid & 31;
  const int f = lane & 7, ig = lane >> 3;  // 4 lane-groups cover i in 32-chunks
  for (int k = wave; k < 128; k += 8) {
    const float* te2k = TE2T + (((size_t)(b * 128 + k)) << 10);  // [i][8]
    float acc = NEGINF;
    const int i0 = ig << 5;
    for (int ii = i0; ii < i0 + 32; ++ii)
      acc = fmaxf(acc, s1[ii][f] + te2k[(ii << 3) + f]);
    acc = fmaxf(acc, __shfl_xor(acc, 8, 32));
    acc = fmaxf(acc, __shfl_xor(acc, 16, 32));
    if (ig == 0) TRIRED[(((size_t)bj << 7) + k) * 8 + f] = acc;
  }
}

// per-edge: te3 on the fly + tri assembly + relu(@W_u3); one wave per edge
__global__ __launch_bounds__(256) void edgeout_kernel(
    const float* __restrict__ E, const int* __restrict__ EI,
    const float* __restrict__ TRIRED, const float* __restrict__ T2,
    const float* __restrict__ T3, const float* __restrict__ TG,
    const float* __restrict__ W_te3, const float* __restrict__ W_u3,
    float* __restrict__ OUT2) {
  const int lane = threadIdx.x & 31;
  const int eidx = blockIdx.x * 8 + (threadIdx.x >> 5);
  const int s0 = EI[eidx], s1v = EI[8192 + eidx];
  const int b = s0 >> 7, j = s0 & 127, k = s1v & 127;
  const float* erow = E + ((((size_t)(b * 128 + j) << 7) + k) << 7);
  float p[8] = {0.f, 0.f, 0.f, 0.f, 0.f, 0.f, 0.f, 0.f};
  for (int c = lane; c < 128; c += 32) {
    const float ev = erow[c];
#pragma unroll
    for (int f = 0; f < 8; ++f) p[f] += ev * W_te3[(c << 3) + f];
  }
#pragma unroll
  for (int off = 16; off > 0; off >>= 1)
#pragma unroll
    for (int f = 0; f < 8; ++f) p[f] += __shfl_xor(p[f], off, 32);
  float vf[8];
#pragma unroll
  for (int f = 0; f < 8; ++f)
    vf[f] = TRIRED[(((size_t)(b * 128 + j) << 7) + k) * 8 + f] +
            T2[((size_t)(b * 128 + j) << 3) + f] +
            T3[((size_t)(b * 128 + k) << 3) + f] + p[f] + TG[(b << 3) + f];
  for (int o = lane; o < 128; o += 32) {
    float acc = 0.f;
#pragma unroll
    for (int f = 0; f < 8; ++f) acc += vf[f] * W_u3[(f << 7) + o];
    OUT2[((size_t)eidx << 7) + o] = fmaxf(acc, 0.f);
  }
}

// ret = LN(z@W_u1 + msgsmax@W_u2) -> OUT1; 4 blocks of 128 rows
__global__ __launch_bounds__(256) void ret_kernel(
    const float* __restrict__ Z, const float* __restrict__ MM,
    const float* __restrict__ W_u1, const float* __restrict__ W_u2,
    const float* __restrict__ lnf_s, const float* __restrict__ lnf_o,
    float* __restrict__ OUT1) {
  __shared__ alignas(16) bf16_t Abf[128][136];
  __shared__ alignas(16) bf16_t Bfr[16384];
  __shared__ alignas(16) float  Cst[128][132];
  __shared__ float redA[256], redB[256];
  const int tid = threadIdx.x;
  const int r0 = blockIdx.x * 128;

  stage_a128(Z + ((size_t)r0 << 7), Abf, tid);
  stage_w_frag<8>(W_u1, Bfr, 128, tid);
  __syncthreads();
  v8f acc[8];
  gemm_slab<true>(Abf, Bfr, acc, tid);
  __syncthreads();

  stage_a128(MM + ((size_t)r0 << 7), Abf, tid);
  stage_w_frag<8>(W_u2, Bfr, 128, tid);
  __syncthreads();
  gemm_slab<false>(Abf, Bfr, acc, tid);  // accumulate second GEMM
  store_tiles(Cst, acc, tid);
  __syncthreads();

  // final LayerNorm (no relu), float4 read/write
  const int row = tid >> 1, fb = (tid & 1) << 6;
  float s = 0.f, ss = 0.f;
#pragma unroll 4
  for (int f = 0; f < 64; f += 4) {
    const float4 x4 = *(const float4*)(&Cst[row][fb + f]);
    s += x4.x + x4.y + x4.z + x4.w;
    ss += x4.x * x4.x + x4.y * x4.y + x4.z * x4.z + x4.w * x4.w;
  }
  redA[tid] = s;
  redB[tid] = ss;
  __syncthreads();
  const float st = redA[tid ^ 1] + s, sst = redB[tid ^ 1] + ss;
  const float mean = st * 0.0078125f;
  const float rstd = rsqrtf(sst * 0.0078125f - mean * mean + 1e-5f);
#pragma unroll 4
  for (int f = 0; f < 64; f += 4) {
    const float4 x4 = *(const float4*)(&Cst[row][fb + f]);
    const float4 s4 = *(const float4*)(lnf_s + fb + f);
    const float4 o4 = *(const float4*)(lnf_o + fb + f);
    float4 y;
    y.x = s4.x * rstd * (x4.x - mean) + o4.x;
    y.y = s4.y * rstd * (x4.y - mean) + o4.y;
    y.z = s4.z * rstd * (x4.z - mean) + o4.z;
    y.w = s4.w * rstd * (x4.w - mean) + o4.w;
    *(float4*)(OUT1 + ((size_t)(r0 + row) << 7) + fb + f) = y;
  }
}

// ------------------------------ launcher ------------------------------------
extern "C" void kernel_launch(void* const* d_in, const int* in_sizes, int n_in,
                              void* d_out, int out_size, void* d_ws, size_t ws_size,
                              hipStream_t stream) {
  (void)in_sizes; (void)n_in; (void)out_size; (void)ws_size;
  const float* node_fts  = (const float*)d_in[0];
  const float* edge_attr = (const float*)d_in[1];
  const float* graph_fts = (const float*)d_in[2];
  const float* W_tri1 = (const float*)d_in[3];
  const float* W_tri2 = (const float*)d_in[4];
  const float* W_tri3 = (const float*)d_in[5];
  const float* W_te1  = (const float*)d_in[6];
  const float* W_te2  = (const float*)d_in[7];
  const float* W_te3  = (const float*)d_in[8];
  const float* W_tg   = (const float*)d_in[9];
  const float* W_m1   = (const float*)d_in[10];
  const float* W_m2   = (const float*)d_in[11];
  const float* W_me   = (const float*)d_in[12];
  const float* W_mg   = (const float*)d_in[13];
  const float* ln1_s  = (const float*)d_in[14];
  const float* ln1_o  = (const float*)d_in[15];
  const float* W_mlp1 = (const float*)d_in[16];
  const float* ln2_s  = (const float*)d_in[17];
  const float* ln2_o  = (const float*)d_in[18];
  const float* W_mlp2 = (const float*)d_in[19];
  const float* W_u1   = (const float*)d_in[20];
  const float* W_u2   = (const float*)d_in[21];
  const float* W_u3   = (const float*)d_in[22];
  const float* lnf_s  = (const float*)d_in[23];
  const float* lnf_o  = (const float*)d_in[24];
  const int*   EI     = (const int*)d_in[25];
  // d_in[26] = batch: implied by edge_index / N for equal-sized graphs

  float* ws     = (float*)d_ws;
  float* E      = ws + OFF_E;
  float* ADJ    = ws + OFF_ADJ;
  float* MSG1   = ws + OFF_MSG1;
  float* MSG2   = ws + OFF_MSG2;
  float* MSGG   = ws + OFF_MSGG;
  float* T1     = ws + OFF_T1;
  float* T2     = ws + OFF_T2;
  float* T3     = ws + OFF_T3;
  float* TG     = ws + OFF_TG;
  float* TE1T   = ws + OFF_TE1T;
  float* TE2T   = ws + OFF_TE2T;
  float* TRIRED = ws + OFF_TRIRED;
  float* MSGF   = ws + OFF_MSGF;
  float* MSGMAX = ws + OFF_MSGMAX;

  float* OUT1 = (float*)d_out;     // [512,128]
  float* OUT2 = OUT1 + 512 * 128;  // [8192,128]

  // 0) zero dense e + adj (contiguous in ws)
  zero_kernel<<<2048, 256, 0, stream>>>((float4*)E, (8388608 + 65536) / 4);
  // 1) scatter edges
  scatter_kernel<<<4096, 256, 0, stream>>>(edge_attr, EI, E, ADJ);
  // 2) small WMMA projections
  gemm_k128_kernel<8><<<4, 256, 0, stream>>>(node_fts, W_m1, MSG1, 512, 128);
  gemm_k128_kernel<8><<<4, 256, 0, stream>>>(node_fts, W_m2, MSG2, 512, 128);
  gemm_k128_kernel<8><<<1, 256, 0, stream>>>(graph_fts, W_mg, MSGG, 4, 128);
  gemm_k128_kernel<1><<<4, 256, 0, stream>>>(node_fts, W_tri1, T1, 512, 8);
  gemm_k128_kernel<1><<<4, 256, 0, stream>>>(node_fts, W_tri2, T2, 512, 8);
  gemm_k128_kernel<1><<<4, 256, 0, stream>>>(node_fts, W_tri3, T3, 512, 8);
  gemm_k128_kernel<1><<<1, 256, 0, stream>>>(graph_fts, W_tg, TG, 4, 8);
  // 3) fused per-(b,i) slab: te1/te2 + 3-GEMM LN/ReLU chain + masked msgs
  mpnn_slab_kernel<<<512, 256, 0, stream>>>(E, ADJ, MSG1, MSG2, MSGG,
                                            W_me, W_mlp1, W_mlp2,
                                            ln1_s, ln1_o, ln2_s, ln2_o,
                                            W_te1, W_te2, TE1T, TE2T, MSGF);
  // 4) max over source nodes
  maxi_kernel<<<512, 256, 0, stream>>>(MSGF, MSGMAX);
  // 5) triplet i-max reduction (never materializes [B,N,N,N,TF])
  trired_kernel<<<512, 256, 0, stream>>>(T1, TE1T, TE2T, TRIRED);
  // 6) per-edge triplet output
  edgeout_kernel<<<1024, 256, 0, stream>>>(E, EI, TRIRED, T2, T3, TG,
                                           W_te3, W_u3, OUT2);
  // 7) ret = LN(z@W_u1 + max@W_u2)
  ret_kernel<<<4, 256, 0, stream>>>(node_fts, MSGMAX, W_u1, W_u2,
                                    lnf_s, lnf_o, OUT1);
}
